// DynamicVAEModelWeak_52261162057786
// MI455X (gfx1250) — matrix operand order repaired and checked by reading, not compiled, not run on hardware
//
#include <hip/hip_runtime.h>
#include <stdint.h>

// ---------------- problem constants ----------------
#define BB   256
#define TPAST 128
#define TFUT  64
#define TT   192          // TPAST + TFUT
#define COVD 64
#define DD   64
#define OUTD 32
#define CTRLD 128
#define HH   512
#define G3   1536         // 3*H
#define BT   (BB*TT)      // 49152 rows, multiple of 16

// ---------------- WMMA types ----------------
typedef __attribute__((ext_vector_type(16))) __bf16 v16bf;
typedef __attribute__((ext_vector_type(8)))  float  v8f;

union AFrag { v16bf v; unsigned u[8]; };

__device__ inline v8f vzero() {
  v8f z;
#pragma unroll
  for (int i = 0; i < 8; ++i) z[i] = 0.f;
  return z;
}

__device__ inline unsigned short f2bf(float f) {
  union { float f; unsigned u; } x;
  x.f = f;
  unsigned u = x.u;
  u += 0x7fffu + ((u >> 16) & 1u);   // round-to-nearest-even
  return (unsigned short)(u >> 16);
}

__device__ inline float sigm(float x) { return 1.f / (1.f + __expf(-x)); }

__device__ inline v8f wmma_bf16(v16bf a, v16bf b, v8f c) {
  // (neg_a, A, neg_b, B, c_mod, C, reuse_a, reuse_b)
  return __builtin_amdgcn_wmma_f32_16x16x32_bf16(false, a, false, b, (short)0, c,
                                                 false, false);
}

// Build 16x32 bf16 A-fragment from an LDS tile [16][strideUS] (bf16/ushort),
// per CDNA5 ISA 16-bit A layout: lane m=L&15, half kh=L>>4,
// VGPR v holds K = (v<4 ? 2v : 8+2v) + 8*kh (+ kt*32), pairs packed per dword.
__device__ inline v16bf load_afrag(const unsigned short* sh, int kt, int lane,
                                   int strideUS) {
  AFrag a;
  const int m = lane & 15;
  const int kh8 = (lane >> 4) << 3;
  const unsigned* p32 = (const unsigned*)sh;
  const int rowbase = m * (strideUS >> 1);   // uint units
#pragma unroll
  for (int v = 0; v < 8; ++v) {
    int kbase = ((v < 4) ? (v << 1) : (8 + (v << 1))) + kh8 + (kt << 5);
    a.u[v] = p32[rowbase + (kbase >> 1)];
  }
  return a.v;
}

// B fragments are pre-swizzled: frag f = nt*KT + kt, lane-major, 16 bf16/lane.
__device__ inline v16bf load_bfrag(const unsigned short* Bf, int f, int lane) {
  return ((const v16bf*)Bf)[(size_t)f * 32 + lane];
}

// ---------------- prep: f32 weight [N][K] -> bf16 B-fragments ----------------
// B-matrix 32x16 layout: lane half h holds K=8h..8h+7 then 16+8h..16+8h+7;
// within a lane, element e maps to k = kbase + (e<8 ? e : e+8).
__global__ void k_prep_frags(const float* __restrict__ W,
                             unsigned short* __restrict__ dst, int N, int K) {
  const int KT = K >> 5;
  int gid = blockIdx.x * blockDim.x + threadIdx.x;   // one thread per (frag,lane)
  int total = (N >> 4) * KT * 32;
  if (gid >= total) return;
  int lane = gid & 31;
  int f = gid >> 5;
  int kt = f % KT, nt = f / KT;
  int n = (nt << 4) + (lane & 15);
  int kb = (kt << 5) + ((lane >> 4) << 3);
  unsigned short* o = dst + (size_t)gid * 16;
  const float* w = W + (size_t)n * K + kb;
#pragma unroll
  for (int e = 0; e < 16; ++e) o[e] = f2bf(w[e < 8 ? e : e + 8]);
}

// ---------------- K1: xg0 = concat(cov) . Wih0^T + bih0 ----------------
// M = BT (row bt = b*TT + t), K = 64, N = 1536. Block = 256 thr = 8 waves.
__global__ __launch_bounds__(256) void k_xg0(const float* __restrict__ past,
                                             const float* __restrict__ fut,
                                             const unsigned short* __restrict__ Wf,
                                             const float* __restrict__ bih,
                                             float* __restrict__ xg) {
  __shared__ __align__(16) unsigned short sA[16 * COVD];
  const int mt = blockIdx.x;
  const int tid = threadIdx.x;
  {  // stage + convert 16x64 f32 -> bf16
    int r = tid >> 4;
    int c4 = (tid & 15) * 4;
    int bt = mt * 16 + r;
    int b = bt / TT, t = bt % TT;
    const float* src = (t < TPAST)
                           ? (past + ((size_t)b * TPAST + t) * COVD + c4)
                           : (fut + ((size_t)b * TFUT + (t - TPAST)) * COVD + c4);
    float4 vv = *(const float4*)src;
    unsigned short* d = sA + r * COVD + c4;
    d[0] = f2bf(vv.x); d[1] = f2bf(vv.y); d[2] = f2bf(vv.z); d[3] = f2bf(vv.w);
  }
  __syncthreads();
  const int lane = tid & 31, w = tid >> 5;
  v8f acc[12];
#pragma unroll
  for (int j = 0; j < 12; ++j) acc[j] = vzero();
#pragma unroll
  for (int kt = 0; kt < 2; ++kt) {
    v16bf a = load_afrag(sA, kt, lane, COVD);
#pragma unroll
    for (int j = 0; j < 12; ++j) {
      int nt = w + j * 8;
      v16bf bm = load_bfrag(Wf, nt * 2 + kt, lane);
      acc[j] = wmma_bf16(a, bm, acc[j]);
    }
  }
  const int nlo = lane & 15, m0 = (lane >> 4) * 8;
#pragma unroll
  for (int j = 0; j < 12; ++j) {
    int col = (w + j * 8) * 16 + nlo;
    float bv = bih[col];
#pragma unroll
    for (int v = 0; v < 8; ++v) {
      size_t row = (size_t)mt * 16 + m0 + v;
      xg[row * G3 + col] = acc[j][v] + bv;
    }
  }
}

// ---------------- generic bf16 GEMM: out[M][Nd] = A[M][512].Wf + bias --------
template <int NT>
__global__ __launch_bounds__(256) void k_gemm_bf16(
    const unsigned short* __restrict__ A, const unsigned short* __restrict__ Wf,
    const float* __restrict__ bias, float* __restrict__ out, int Nd) {
  constexpr int NPW = (NT + 7) / 8;
  __shared__ __align__(16) unsigned short sA[16 * HH];
  const int mt = blockIdx.x;
  const int tid = threadIdx.x;
  {  // tile rows are contiguous: 16*512 bf16 = 1024 uint4
    const uint4* src = (const uint4*)(A + (size_t)mt * 16 * HH);
    uint4* dst = (uint4*)sA;
#pragma unroll
    for (int s = 0; s < 4; ++s) dst[tid + s * 256] = src[tid + s * 256];
  }
  __syncthreads();
  const int lane = tid & 31, w = tid >> 5;
  v8f acc[NPW];
#pragma unroll
  for (int j = 0; j < NPW; ++j) acc[j] = vzero();
#pragma unroll 4
  for (int kt = 0; kt < 16; ++kt) {
    v16bf a = load_afrag(sA, kt, lane, HH);
#pragma unroll
    for (int j = 0; j < NPW; ++j) {
      int nt = w + j * 8;
      if (nt < NT) {
        v16bf bm = load_bfrag(Wf, nt * 16 + kt, lane);
        acc[j] = wmma_bf16(a, bm, acc[j]);
      }
    }
  }
  const int nlo = lane & 15, m0 = (lane >> 4) * 8;
#pragma unroll
  for (int j = 0; j < NPW; ++j) {
    int nt = w + j * 8;
    if (nt < NT) {
      int col = nt * 16 + nlo;
      float bv = bias[col];
#pragma unroll
      for (int v = 0; v < 8; ++v) {
        size_t row = (size_t)mt * 16 + m0 + v;
        out[row * Nd + col] = acc[j][v] + bv;
      }
    }
  }
}

// ---------------- GRU recurrence: one block per 16-row batch tile ------------
// xg: [B][T][1536] f32 (precomputed input gates), Wf: Whh fragments (KT=16),
// bhh: [1536], hout: [B][T][512] bf16. 512 threads = 16 waves; wave w owns
// hidden-column groups [16w,16w+16) and [16w+256,16w+272) -> n-tiles
// {w, w+32, w+64} and {w+16, w+48, w+80} (r/z/n gates are wave-local).
__global__ __launch_bounds__(512) void k_gru(const float* __restrict__ xg,
                                             const unsigned short* __restrict__ Wf,
                                             const float* __restrict__ bhh,
                                             unsigned short* __restrict__ hout,
                                             int T) {
  __shared__ __align__(16) unsigned short hbf[16 * HH];  // bf16 h (A-operand)
  __shared__ float hf[16 * HH];                          // f32 carry
  const int tid = threadIdx.x;
  const int lane = tid & 31, w = tid >> 5;
  const int btile = blockIdx.x;
  for (int i = tid; i < 16 * HH; i += 512) { hbf[i] = 0; hf[i] = 0.f; }
  __syncthreads();
  const int nlo = lane & 15, m0 = (lane >> 4) * 8;
  const int colA = (w << 4) + nlo;
  const int colB = colA + 256;
  const float br0 = bhh[colA], bz0 = bhh[colA + 512], bn0 = bhh[colA + 1024];
  const float br1 = bhh[colB], bz1 = bhh[colB + 512], bn1 = bhh[colB + 1024];

  for (int t = 0; t < T; ++t) {
    v8f aR0 = vzero(), aZ0 = vzero(), aN0 = vzero();
    v8f aR1 = vzero(), aZ1 = vzero(), aN1 = vzero();
#pragma unroll 4
    for (int kt = 0; kt < 16; ++kt) {
      v16bf a = load_afrag(hbf, kt, lane, HH);
      v16bf b0 = load_bfrag(Wf, (w +  0) * 16 + kt, lane);
      v16bf b1 = load_bfrag(Wf, (w + 32) * 16 + kt, lane);
      v16bf b2 = load_bfrag(Wf, (w + 64) * 16 + kt, lane);
      v16bf b3 = load_bfrag(Wf, (w + 16) * 16 + kt, lane);
      v16bf b4 = load_bfrag(Wf, (w + 48) * 16 + kt, lane);
      v16bf b5 = load_bfrag(Wf, (w + 80) * 16 + kt, lane);
      aR0 = wmma_bf16(a, b0, aR0);
      aZ0 = wmma_bf16(a, b1, aZ0);
      aN0 = wmma_bf16(a, b2, aN0);
      aR1 = wmma_bf16(a, b3, aR1);
      aZ1 = wmma_bf16(a, b4, aZ1);
      aN1 = wmma_bf16(a, b5, aN1);
    }
    __syncthreads();  // all A-fragment reads of old h complete
#pragma unroll
    for (int v = 0; v < 8; ++v) {
      int m = m0 + v;
      size_t bglob = (size_t)btile * 16 + m;
      size_t xb = (bglob * T + t) * G3;
      {  // group 0
        float xr = xg[xb + colA], xz = xg[xb + colA + 512],
              xn = xg[xb + colA + 1024];
        float r = sigm(xr + aR0[v] + br0);
        float z = sigm(xz + aZ0[v] + bz0);
        float nn = tanhf(xn + r * (aN0[v] + bn0));
        float hold = hf[m * HH + colA];
        float hnew = (1.f - z) * nn + z * hold;
        hf[m * HH + colA] = hnew;
        unsigned short hb = f2bf(hnew);
        hbf[m * HH + colA] = hb;
        hout[(bglob * T + t) * HH + colA] = hb;
      }
      {  // group 1
        float xr = xg[xb + colB], xz = xg[xb + colB + 512],
              xn = xg[xb + colB + 1024];
        float r = sigm(xr + aR1[v] + br1);
        float z = sigm(xz + aZ1[v] + bz1);
        float nn = tanhf(xn + r * (aN1[v] + bn1));
        float hold = hf[m * HH + colB];
        float hnew = (1.f - z) * nn + z * hold;
        hf[m * HH + colB] = hnew;
        unsigned short hb = f2bf(hnew);
        hbf[m * HH + colB] = hb;
        hout[(bglob * T + t) * HH + colB] = hb;
      }
    }
    __syncthreads();  // h fully updated before next step's fragment gathers
  }
}

// ---------------- sequential filter (VALU; 4 batch rows / block) -------------
__global__ __launch_bounds__(256) void k_filter(
    const float* __restrict__ feat,        // [B][T][64] = cov_feat.featW^T + b
    const float* __restrict__ past_label,  // [B][TPAST][32]
    const float* __restrict__ flow_u, const float* __restrict__ flow_w,
    const float* __restrict__ flow_b, const float* __restrict__ ctrl_W1,
    const float* __restrict__ ctrl_b1, const float* __restrict__ ctrl_W2,
    const float* __restrict__ ctrl_b2, const float* __restrict__ ctrlmu_W,
    const float* __restrict__ ctrlmu_b, const float* __restrict__ out_W,
    const float* __restrict__ out_b, float* __restrict__ out) {
  __shared__ float sMu[4][DD], sCtl[4][OUTD], sHdn[4][CTRLD];
  __shared__ float sCm[4][DD], sCs[4][DD];
  const int tid = threadIdx.x;
  const int r = tid >> 6, d = tid & 63;
  const int b = blockIdx.x * 4 + r;
  const float fu = flow_u[d], fw = flow_w[d], fb = flow_b[d];
  float mu = 0.f, var = 1.f;
  float* predOut = out;
  float* cmOut = out + (size_t)BB * TT * OUTD;
  float* csOut = cmOut + (size_t)BB * (TT - 1) * DD;

  for (int t = 0; t < TT; ++t) {
    float a = tanhf(fw * mu + fb);
    float mu_pr = fu * a + feat[((size_t)b * TT + t) * DD + d] + mu;
    float coef = fu * fw * (1.f - a * a);
    float var_pr = 2.f * coef * var + var + 1.f;
    sMu[r][d] = mu_pr;
    __syncthreads();
    if (tid < 128) {  // pred + control input: 4 rows x 32 outputs
      int rr = tid >> 5, o = tid & 31;
      float s = out_b[o];
      const float* wrow = out_W + o * DD;
#pragma unroll 8
      for (int k = 0; k < DD; ++k) s += sMu[rr][k] * wrow[k];
      int bg = blockIdx.x * 4 + rr;
      predOut[((size_t)bg * TT + t) * OUTD + o] = s;
      sCtl[rr][o] =
          (t < TPAST) ? past_label[((size_t)bg * TPAST + t) * OUTD + o] : s;
    }
    __syncthreads();
#pragma unroll
    for (int ii = 0; ii < 2; ++ii) {  // hdn: 4 rows x 128
      int i = tid + ii * 256;
      int rr = i >> 7, c = i & 127;
      float s = ctrl_b1[c];
      const float* wrow = ctrl_W1 + c * OUTD;
#pragma unroll 8
      for (int k = 0; k < OUTD; ++k) s += sCtl[rr][k] * wrow[k];
      sHdn[rr][c] = tanhf(s);
    }
    __syncthreads();
#pragma unroll
    for (int ii = 0; ii < 2; ++ii) {  // g2 -> cm / cs
      int i = tid + ii * 256;
      int rr = i >> 7, j = i & 127;
      float s = ctrl_b2[j];
      const float* wrow = ctrl_W2 + j * CTRLD;
#pragma unroll 8
      for (int k = 0; k < CTRLD; ++k) s += sHdn[rr][k] * wrow[k];
      int bg = blockIdx.x * 4 + rr;
      if (j < DD) {
        float mm = ctrlmu_b[j];
        const float* mrow = ctrlmu_W + j * DD;
#pragma unroll 8
        for (int k = 0; k < DD; ++k) mm += sMu[rr][k] * mrow[k];
        float cmv = s - mm;
        sCm[rr][j] = cmv;
        if (t > 0) cmOut[((size_t)bg * (TT - 1) + (t - 1)) * DD + j] = cmv;
      } else {
        int j2 = j - DD;
        float csv = fmaxf(s, 0.f) + log1pf(__expf(-fabsf(s)));  // softplus
        sCs[rr][j2] = csv;
        if (t > 0) csOut[((size_t)bg * (TT - 1) + (t - 1)) * DD + j2] = csv;
      }
    }
    __syncthreads();
    float var_po = sCs[r][d], mu_po = sCm[r][d];
    float ssum = var_po + var_pr;
    mu = (var_po / ssum) * mu_pr + (var_pr / ssum) * mu_po;
    var = var_pr * var_po / ssum;
    __syncthreads();
  }
}

// ---------------- host ----------------
extern "C" void kernel_launch(void* const* d_in, const int* in_sizes, int n_in,
                              void* d_out, int out_size, void* d_ws,
                              size_t ws_size, hipStream_t stream) {
  (void)in_sizes; (void)n_in; (void)out_size; (void)ws_size;
  const float* past_input   = (const float*)d_in[0];
  const float* past_label   = (const float*)d_in[1];
  const float* future_input = (const float*)d_in[2];
  const float* Wih0 = (const float*)d_in[3];
  const float* Whh0 = (const float*)d_in[4];
  const float* bih0 = (const float*)d_in[5];
  const float* bhh0 = (const float*)d_in[6];
  const float* Wih1 = (const float*)d_in[7];
  const float* Whh1 = (const float*)d_in[8];
  const float* bih1 = (const float*)d_in[9];
  const float* bhh1 = (const float*)d_in[10];
  const float* flow_u = (const float*)d_in[11];
  const float* flow_w = (const float*)d_in[12];
  const float* flow_b = (const float*)d_in[13];
  const float* featW  = (const float*)d_in[14];
  const float* featb  = (const float*)d_in[15];
  const float* ctrlW1 = (const float*)d_in[16];
  const float* ctrlb1 = (const float*)d_in[17];
  const float* ctrlW2 = (const float*)d_in[18];
  const float* ctrlb2 = (const float*)d_in[19];
  const float* cmuW   = (const float*)d_in[20];
  const float* cmub   = (const float*)d_in[21];
  const float* outW   = (const float*)d_in[22];
  const float* outb   = (const float*)d_in[23];

  char* ws = (char*)d_ws;
  size_t off = 0;
  auto nxt = [&](size_t bytes) {
    size_t o = off;
    off += (bytes + 255) & ~(size_t)255;
    return o;
  };
  float* xg = (float*)(ws + nxt((size_t)BT * G3 * 4));              // reused L0/L1
  unsigned short* h1 = (unsigned short*)(ws + nxt((size_t)BT * HH * 2));
  unsigned short* cf = (unsigned short*)(ws + nxt((size_t)BT * HH * 2));
  float* feat = (float*)(ws + nxt((size_t)BT * DD * 4));
  unsigned short* fWih0 = (unsigned short*)(ws + nxt((size_t)96 * 2 * 512 * 2));
  unsigned short* fWhh0 = (unsigned short*)(ws + nxt((size_t)96 * 16 * 512 * 2));
  unsigned short* fWih1 = (unsigned short*)(ws + nxt((size_t)96 * 16 * 512 * 2));
  unsigned short* fWhh1 = (unsigned short*)(ws + nxt((size_t)96 * 16 * 512 * 2));
  unsigned short* fFeat = (unsigned short*)(ws + nxt((size_t)4 * 16 * 512 * 2));

  auto prep = [&](const float* W, unsigned short* dst, int N, int K) {
    int total = (N / 16) * (K / 32) * 32;
    k_prep_frags<<<(total + 255) / 256, 256, 0, stream>>>(W, dst, N, K);
  };
  prep(Wih0, fWih0, G3, COVD);
  prep(Whh0, fWhh0, G3, HH);
  prep(Wih1, fWih1, G3, HH);
  prep(Whh1, fWhh1, G3, HH);
  prep(featW, fFeat, DD, HH);

  // xg0 = concat(past,future) . Wih0^T + bih0   (time-parallel)
  k_xg0<<<BT / 16, 256, 0, stream>>>(past_input, future_input, fWih0, bih0, xg);
  // layer-0 recurrence -> h1 (bf16)
  k_gru<<<16, 512, 0, stream>>>(xg, fWhh0, bhh0, h1, TT);
  // xg1 = h1 . Wih1^T + bih1   (time-parallel, reuses xg buffer)
  k_gemm_bf16<96><<<BT / 16, 256, 0, stream>>>(h1, fWih1, bih1, xg, G3);
  // layer-1 recurrence -> cov_feat (bf16)
  k_gru<<<16, 512, 0, stream>>>(xg, fWhh1, bhh1, cf, TT);
  // feat = cov_feat . featW^T + feat_b   (time-parallel)
  k_gemm_bf16<4><<<BT / 16, 256, 0, stream>>>(cf, fFeat, featb, feat, DD);
  // sequential probabilistic filter -> pred / cm / cs
  k_filter<<<BB / 4, 256, 0, stream>>>(feat, past_label, flow_u, flow_w, flow_b,
                                       ctrlW1, ctrlb1, ctrlW2, ctrlb2, cmuW,
                                       cmub, outW, outb, (float*)d_out);
}
